// MGFNet_60138132078625
// MI455X (gfx1250) — matrix operand order repaired and checked by reading, not compiled, not run on hardware
//
#include <hip/hip_runtime.h>

// ---------------------------------------------------------------------------
// MGFNet forward on MI455X (gfx1250): all convs + covariance pooling run as
// implicit GEMM on v_wmma_f32_16x16x32_bf16 (bf16 A/B, f32 accumulate).
// Block = 256 threads = 8 wave32s; block tile = 64 Co x 64 px; each wave owns
// one A fragment (16 co x 32 k) reused across TWO 16x16 output tiles
// (2 WMMAs per K-chunk per wave).  All offset arithmetic is 32-bit (every
// tensor < 2^31 elements) and loop-invariant terms are hoisted.
// ---------------------------------------------------------------------------

typedef __attribute__((ext_vector_type(16))) __bf16        v16bf;
typedef __attribute__((ext_vector_type(8)))  float         v8f;
typedef __attribute__((ext_vector_type(4)))  unsigned int  v4u;

union FragBF { v16bf v; v4u q[2]; };

#define ACT_NONE 0
#define ACT_RELU 1
#define ACT_SIG  2

__device__ __forceinline__ unsigned short f32bf(float f) {
  unsigned int u = __builtin_bit_cast(unsigned int, f);
  u += 0x7fffu + ((u >> 16) & 1u);           // round-to-nearest-even
  return (unsigned short)(u >> 16);
}

__device__ __forceinline__ float epilogue(float y, int co,
                                          const float* bias, const float* gamma,
                                          const float* beta, int act, int post) {
  if (bias)  y += bias[co];
  if (!post && gamma) y = y * gamma[co] + beta[co];
  if (act == ACT_RELU)      y = fmaxf(y, 0.0f);
  else if (act == ACT_SIG)  y = 1.0f / (1.0f + __expf(-y));
  if (post && gamma) y = y * gamma[co] + beta[co];
  return y;
}

// ---------------------------------------------------------------------------
// Generic conv (KHxKW, stride/pad/dilation) as implicit GEMM with WMMA.
// in  : [2, CiTot, H, W]  (pointer may be pre-offset to a channel slice;
//                          in_b_stride = CiTot*H*W covers the full tensor)
// wgt : [Co, Ci, KH, KW]
// out : [2, CoTot, Ho, Wo] (pointer may be pre-offset; out_b_stride full)
// ---------------------------------------------------------------------------
__global__ __launch_bounds__(256)
void wmma_conv_kernel(const float* __restrict__ in, const float* __restrict__ wgt,
                      const float* __restrict__ bias, const float* __restrict__ gamma,
                      const float* __restrict__ beta, float* __restrict__ out,
                      int Ci, int Co, int H, int W, int Ho, int Wo,
                      int KH, int KW, int stride, int pad, int dil,
                      int in_b_stride, int out_b_stride, int act, int post)
{
  __shared__ __align__(16) unsigned short ldsA[64][40];  // [co][k] bf16, 80B rows
  __shared__ __align__(16) unsigned short ldsB[64][40];  // [px][k] bf16

  const int tid  = threadIdx.x;
  const int lane = tid & 31;
  const int wv   = tid >> 5;
  const int b    = blockIdx.z;
  const int P    = Ho * Wo;
  const int pxTile = blockIdx.x * 64;
  const int coTile = blockIdx.y * 64;

  // wave tile: co sub-tile = wv&3 (16 rows), px pair = wv>>2 (two 16-col tiles)
  const int coSub  = (wv & 3) << 4;
  const int pxSub0 = (wv >> 2) << 5;
  const int mrow   = lane & 15;
  const int khalf  = lane >> 4;

  v8f acc0 = {0.f,0.f,0.f,0.f,0.f,0.f,0.f,0.f};
  v8f acc1 = {0.f,0.f,0.f,0.f,0.f,0.f,0.f,0.f};

  // staging constants (per thread, loop invariant)
  const int a_kl  = tid & 31;   // A: ci within chunk
  const int a_cl0 = tid >> 5;   // A: co row base, +8 per t
  const int b_pl  = tid & 63;   // B: pixel within tile
  const int b_kl0 = tid >> 6;   // B: ci base, +4 per t

  const int p_g    = pxTile + b_pl;
  const bool pValid = p_g < P;
  const int oy  = pValid ? (p_g / Wo) : 0;
  const int ox  = pValid ? (p_g - oy * Wo) : 0;
  const int iy0 = oy * stride - pad;
  const int ix0 = ox * stride - pad;

  const int HW      = H * W;
  const int KHW     = KH * KW;
  const int inBatch = b * in_b_stride;
  const int nck     = (Ci + 31) >> 5;

  for (int tap = 0; tap < KHW; ++tap) {
    const int kh = tap / KW, kw = tap - kh * KW;
    const int iy = iy0 + kh * dil;
    const int ix = ix0 + kw * dil;
    const bool inB = pValid && iy >= 0 && iy < H && ix >= 0 && ix < W;
    const int pixOff = inBatch + iy * W + ix;   // add ci*HW per element
    const int wTap   = kh * KW + kw;

    for (int ck = 0; ck < nck; ++ck) {
      const int ciBase = ck << 5;
      __syncthreads();
      // ---- stage A: 64 co x 32 k weights (8 elems/thread) ----
      {
        const int ci   = ciBase + a_kl;
        const bool ciOK = ci < Ci;
        const int wOff = ci * KHW + wTap;
#pragma unroll
        for (int t = 0; t < 8; ++t) {
          const int cl = a_cl0 + (t << 3);
          const int co = coTile + cl;
          float v = 0.0f;
          if (ciOK && co < Co) v = wgt[co * (Ci * KHW) + wOff];
          ldsA[cl][a_kl] = f32bf(v);
        }
      }
      // ---- stage B: 64 px x 32 k im2col patch (8 elems/thread, px coalesced) ----
#pragma unroll
      for (int t = 0; t < 8; ++t) {
        const int kl = b_kl0 + (t << 2);
        const int ci = ciBase + kl;
        float v = 0.0f;
        if (inB && ci < Ci) v = in[pixOff + ci * HW];
        ldsB[b_pl][kl] = f32bf(v);
      }
      // prefetch next Ci chunk into L2 (global_prefetch_b8)
      if (ck + 1 < nck) {
        const int cin = ciBase + 32 + b_kl0;
        if (inB && cin < Ci) __builtin_prefetch(&in[pixOff + cin * HW], 0, 1);
      }
      __syncthreads();
      // ---- fragments A(16x32)/B(32x16): contiguous k = 16*khalf + s, same order for A and B ----
      FragBF fa, fb0, fb1;
      const unsigned short* ra = &ldsA[coSub + mrow][0];
      fa.q[0] = *reinterpret_cast<const v4u*>(ra + (khalf << 4));
      fa.q[1] = *reinterpret_cast<const v4u*>(ra + (khalf << 4) + 8);
      const unsigned short* rb0 = &ldsB[pxSub0 + mrow][0];
      fb0.q[0] = *reinterpret_cast<const v4u*>(rb0 + (khalf << 4));
      fb0.q[1] = *reinterpret_cast<const v4u*>(rb0 + (khalf << 4) + 8);
      const unsigned short* rb1 = &ldsB[pxSub0 + 16 + mrow][0];
      fb1.q[0] = *reinterpret_cast<const v4u*>(rb1 + (khalf << 4));
      fb1.q[1] = *reinterpret_cast<const v4u*>(rb1 + (khalf << 4) + 8);
      acc0 = __builtin_amdgcn_wmma_f32_16x16x32_bf16(false, fa.v, false, fb0.v,
                                                     (short)0, acc0, false, false);
      acc1 = __builtin_amdgcn_wmma_f32_16x16x32_bf16(false, fa.v, false, fb1.v,
                                                     (short)0, acc1, false, false);
    }
  }
  // ---- epilogue: C/D layout => element (r + 8*(lane>>4), lane&15) ----
  const int col = lane & 15;
  const int hi  = lane >> 4;
  const int outBatch = b * out_b_stride;
#pragma unroll
  for (int r = 0; r < 8; ++r) {
    const int co = coTile + coSub + r + (hi << 3);
    if (co < Co) {
      const int p0 = pxTile + pxSub0 + col;
      const int p1 = p0 + 16;
      if (p0 < P) out[outBatch + co * P + p0] =
          epilogue(acc0[r], co, bias, gamma, beta, act, post);
      if (p1 < P) out[outBatch + co * P + p1] =
          epilogue(acc1[r], co, bias, gamma, beta, act, post);
    }
  }
}

// ---------------------------------------------------------------------------
// Covariance pooling: cov[b,c,d] = row_g[c]*( (1/M) sum_m xc[c,m]*xc[d,m] ) + row_b[c]
// xc = a - mean (centered on load). 64x64 block tile, 2 WMMAs/wave per chunk;
// row bases + means hoisted to registers (inner loop runs M/32 = 2048 times).
// ---------------------------------------------------------------------------
__global__ __launch_bounds__(256)
void wmma_covpool_kernel(const float* __restrict__ a, const float* __restrict__ mean,
                         const float* __restrict__ rg, const float* __restrict__ rb,
                         float* __restrict__ cov, int C, int M)
{
  __shared__ __align__(16) unsigned short ldsA[64][40];
  __shared__ __align__(16) unsigned short ldsB[64][40];

  const int tid  = threadIdx.x;
  const int lane = tid & 31;
  const int wv   = tid >> 5;
  const int b    = blockIdx.z;
  const int dTile = blockIdx.x * 64;
  const int rTile = blockIdx.y * 64;

  const int rSub  = (wv & 3) << 4;
  const int dSub0 = (wv >> 2) << 5;
  const int mrow  = lane & 15;
  const int khalf = lane >> 4;

  v8f acc0 = {0.f,0.f,0.f,0.f,0.f,0.f,0.f,0.f};
  v8f acc1 = {0.f,0.f,0.f,0.f,0.f,0.f,0.f,0.f};

  const int ml  = tid & 31;
  const int rl0 = tid >> 5;
  int   baseR[8], baseD[8];
  float muR[8],  muD[8];
#pragma unroll
  for (int t = 0; t < 8; ++t) {
    const int r = min(rTile + rl0 + (t << 3), C - 1);
    const int d = min(dTile + rl0 + (t << 3), C - 1);
    baseR[t] = (b * C + r) * M + ml;
    baseD[t] = (b * C + d) * M + ml;
    muR[t] = mean[b * C + r];
    muD[t] = mean[b * C + d];
  }

  for (int mk = 0; mk < M; mk += 32) {
    __syncthreads();
#pragma unroll
    for (int t = 0; t < 8; ++t)
      ldsA[rl0 + (t << 3)][ml] = f32bf(a[baseR[t] + mk] - muR[t]);
#pragma unroll
    for (int t = 0; t < 8; ++t)
      ldsB[rl0 + (t << 3)][ml] = f32bf(a[baseD[t] + mk] - muD[t]);
    __syncthreads();
    FragBF fa, fb0, fb1;
    const unsigned short* ra = &ldsA[rSub + mrow][0];
    fa.q[0] = *reinterpret_cast<const v4u*>(ra + (khalf << 4));
    fa.q[1] = *reinterpret_cast<const v4u*>(ra + (khalf << 4) + 8);
    const unsigned short* rb0 = &ldsB[dSub0 + mrow][0];
    fb0.q[0] = *reinterpret_cast<const v4u*>(rb0 + (khalf << 4));
    fb0.q[1] = *reinterpret_cast<const v4u*>(rb0 + (khalf << 4) + 8);
    const unsigned short* rb1 = &ldsB[dSub0 + 16 + mrow][0];
    fb1.q[0] = *reinterpret_cast<const v4u*>(rb1 + (khalf << 4));
    fb1.q[1] = *reinterpret_cast<const v4u*>(rb1 + (khalf << 4) + 8);
    acc0 = __builtin_amdgcn_wmma_f32_16x16x32_bf16(false, fa.v, false, fb0.v,
                                                   (short)0, acc0, false, false);
    acc1 = __builtin_amdgcn_wmma_f32_16x16x32_bf16(false, fa.v, false, fb1.v,
                                                   (short)0, acc1, false, false);
  }
  const int col = lane & 15;
  const int hi  = lane >> 4;
  const float invM = 1.0f / (float)M;
#pragma unroll
  for (int r = 0; r < 8; ++r) {
    const int c = rTile + rSub + r + (hi << 3);
    if (c < C) {
      const float s = invM * rg[c];
      const int d0 = dTile + dSub0 + col;
      const int d1 = d0 + 16;
      if (d0 < C) cov[(b * C + c) * C + d0] = acc0[r] * s + rb[c];
      if (d1 < C) cov[(b * C + c) * C + d1] = acc1[r] * s + rb[c];
    }
  }
}

// ---------------- small helper kernels (bandwidth-bound) -------------------

__global__ void mean_hw_kernel(const float* __restrict__ in, float* __restrict__ mean, int M) {
  __shared__ float red[256];
  long base = (long)blockIdx.x * M;
  float s = 0.0f;
  for (int i = threadIdx.x; i < M; i += 256) s += in[base + i];
  red[threadIdx.x] = s; __syncthreads();
  for (int o = 128; o > 0; o >>= 1) {
    if (threadIdx.x < o) red[threadIdx.x] += red[threadIdx.x + o];
    __syncthreads();
  }
  if (threadIdx.x == 0) mean[blockIdx.x] = red[0] / (float)M;
}

// rc[b, c*4+k] = sum_r cov[b,c,r] * rc_w[c,k,r] + rc_b[c,k]
__global__ void rc_kernel(const float* __restrict__ cov, const float* __restrict__ w,
                          const float* __restrict__ bias, float* __restrict__ rc, int C) {
  __shared__ float red[128];
  int bc = blockIdx.x, b = bc / C, c = bc - b * C;
  int k = threadIdx.x >> 5, lane = threadIdx.x & 31;
  float s = 0.0f;
  for (int r = lane; r < C; r += 32)
    s += cov[bc * C + r] * w[(c * 4 + k) * C + r];
  red[threadIdx.x] = s; __syncthreads();
  for (int o = 16; o > 0; o >>= 1) {
    if (lane < o) red[threadIdx.x] += red[threadIdx.x + o];
    __syncthreads();
  }
  if (lane == 0) rc[b * 4 * C + c * 4 + k] = red[threadIdx.x] + bias[c * 4 + k];
}

__global__ void scale_att_kernel(const float* __restrict__ x, const float* __restrict__ att,
                                 float* __restrict__ out, int C, long P, long total) {
  long idx = (long)blockIdx.x * blockDim.x + threadIdx.x;
  if (idx >= total) return;
  long b = idx / ((long)C * P);
  long c = (idx / P) % C;
  out[idx] = x[idx] * att[b * C + c];
}

__global__ void add_kernel(const float* __restrict__ A, long as,
                           const float* __restrict__ Bp, long bs,
                           float* __restrict__ O, long os, long perB) {
  long idx = (long)blockIdx.x * blockDim.x + threadIdx.x;
  if (idx >= 2 * perB) return;
  long b = idx / perB, i = idx - b * perB;
  O[b * os + i] = A[b * as + i] + Bp[b * bs + i];
}

// bilinear x2 upsample, align_corners=False (half-pixel centers, edge clamp)
__global__ void up2_kernel(const float* __restrict__ in, float* __restrict__ out,
                           int C, int H, int W, long total) {
  long idx = (long)blockIdx.x * blockDim.x + threadIdx.x;
  if (idx >= total) return;
  int Ho = 2 * H, Wo = 2 * W;
  long PW = (long)Ho * Wo;
  long bc = idx / PW;
  long p  = idx - bc * PW;
  int oy = (int)(p / Wo), ox = (int)(p - (long)oy * Wo);
  float sy = oy * 0.5f - 0.25f, sx = ox * 0.5f - 0.25f;
  int y0 = (int)floorf(sy), x0 = (int)floorf(sx);
  float wy = sy - y0, wx = sx - x0;
  int y0c = min(max(y0, 0), H - 1), y1c = min(max(y0 + 1, 0), H - 1);
  int x0c = min(max(x0, 0), W - 1), x1c = min(max(x0 + 1, 0), W - 1);
  const float* src = in + bc * (long)H * W;
  float v = (1 - wy) * ((1 - wx) * src[y0c * W + x0c] + wx * src[y0c * W + x1c])
          +      wy  * ((1 - wx) * src[y1c * W + x0c] + wx * src[y1c * W + x1c]);
  out[idx] = v;
}

// GFF combine + concat: cat[:,0:C]=(1+g1)x1+(1-g1)(g2*x2u); cat[:,C:2C]=(1+g2)x2u+(1-g2)(g1*x1)
__global__ void gff_kernel(const float* __restrict__ x1, const float* __restrict__ x2u,
                           const float* __restrict__ g1, const float* __restrict__ g2,
                           float* __restrict__ cat, int C, long P, long total) {
  long idx = (long)blockIdx.x * blockDim.x + threadIdx.x;
  if (idx >= total) return;
  long perB = 2L * C * P;
  long b = idx / perB;
  long rem = idx - b * perB;
  long ch = rem / P;
  long p  = rem - ch * P;
  float G1 = g1[b * P + p], G2 = g2[b * P + p];
  long i1 = (b * C + (ch % C)) * P + p;
  float v1 = x1[i1], v2 = x2u[i1];
  float r = (ch < C) ? (1.0f + G1) * v1 + (1.0f - G1) * (G2 * v2)
                     : (1.0f + G2) * v2 + (1.0f - G2) * (G1 * v1);
  cat[idx] = r;
}

// EMFF avg branch: write relu(bn(mean)) broadcast into channel slice of xb
__global__ void bcast_avg_kernel(const float* __restrict__ avg, const float* __restrict__ g,
                                 const float* __restrict__ bt, float* __restrict__ out,
                                 long obs, int C, int P, long total) {
  long idx = (long)blockIdx.x * blockDim.x + threadIdx.x;
  if (idx >= total) return;
  long b = idx / ((long)C * P);
  long rem = idx - b * (long)C * P;
  int c = (int)(rem / P);
  long p = rem - (long)c * P;
  float v = avg[b * C + c] * g[c] + bt[c];
  out[b * obs + (long)c * P + p] = fmaxf(v, 0.0f);
}

// ---------------------------------------------------------------------------

extern "C" void kernel_launch(void* const* d_in, const int* in_sizes, int n_in,
                              void* d_out, int out_size, void* d_ws, size_t ws_size,
                              hipStream_t stream) {
  (void)in_sizes; (void)n_in; (void)out_size; (void)ws_size;

  const float* X = (const float*)d_in[0];       // [2,103,256,256]
  int ii = 1;
  auto nxt = [&]() { return (const float*)d_in[ii++]; };
  // params in setup_inputs() insertion order
  const float *gsa_c1_w = nxt(), *gsa_bn1_g = nxt(), *gsa_bn1_b = nxt();
  const float *gsa_dr_w = nxt(), *gsa_dr_b = nxt(), *gsa_dr_g = nxt(), *gsa_dr_bb = nxt();
  const float *gsa_row_g = nxt(), *gsa_row_b = nxt();
  const float *gsa_rc_w = nxt(), *gsa_rc_b = nxt();
  const float *gsa_fc_w = nxt(), *gsa_fc_b = nxt();
  const float *c1_w = nxt(), *c1_b = nxt(), *c1_g = nxt(), *c1_bb = nxt();
  const float *c2_w = nxt(), *c2_b = nxt(), *c2_g = nxt(), *c2_bb = nxt();
  const float *c3_w = nxt(), *c3_b = nxt(), *c3_g = nxt(), *c3_bb = nxt();
  const float *c4_w = nxt(), *c4_b = nxt(), *c4_g = nxt(), *c4_bb = nxt();
  const float *c5_w = nxt(), *c5_b = nxt(), *c5_g = nxt(), *c5_bb = nxt();
  const float *g1_c1_w = nxt(), *g1_c1_g = nxt(), *g1_c1_b = nxt();
  const float *g1_c2_w = nxt(), *g1_c2_g = nxt(), *g1_c2_b = nxt();
  const float *g1_a_w1 = nxt(), *g1_a_b1 = nxt(), *g1_a_g = nxt(), *g1_a_bb = nxt();
  const float *g1_a_w2 = nxt(), *g1_a_b2 = nxt(), *g1_a_g2 = nxt(), *g1_a_bb2 = nxt();
  const float *g1_b_w1 = nxt(), *g1_b_b1 = nxt(), *g1_b_g = nxt(), *g1_b_bb = nxt();
  const float *g1_b_w2 = nxt(), *g1_b_b2 = nxt(), *g1_b_g2 = nxt(), *g1_b_bb2 = nxt();
  const float *g1_p_w = nxt(), *g1_p_g = nxt(), *g1_p_b = nxt();
  const float *g2_c1_w = nxt(), *g2_c1_g = nxt(), *g2_c1_b = nxt();
  const float *g2_c2_w = nxt(), *g2_c2_g = nxt(), *g2_c2_b = nxt();
  const float *g2_a_w1 = nxt(), *g2_a_b1 = nxt(), *g2_a_g = nxt(), *g2_a_bb = nxt();
  const float *g2_a_w2 = nxt(), *g2_a_b2 = nxt(), *g2_a_g2 = nxt(), *g2_a_bb2 = nxt();
  const float *g2_b_w1 = nxt(), *g2_b_b1 = nxt(), *g2_b_g = nxt(), *g2_b_bb = nxt();
  const float *g2_b_w2 = nxt(), *g2_b_b2 = nxt(), *g2_b_g2 = nxt(), *g2_b_bb2 = nxt();
  const float *g2_p_w = nxt(), *g2_p_g = nxt(), *g2_p_b = nxt();
  const float *e1_w = nxt(), *e1_g = nxt(), *e1_b = nxt();
  const float *e2_w = nxt(), *e2_g = nxt(), *e2_b = nxt();
  const float *e3_w = nxt(), *e3_g = nxt(), *e3_b = nxt();
  const float *eavg_g = nxt(), *eavg_b = nxt();
  const float *last_w = nxt(), *last_b = nxt();

  // ---- workspace plan (bump alloc, 256B granularity; peak ~382 MB) ----
  size_t off = 0;
  auto alloc = [&](long elems) {
    float* p = (float*)((char*)d_ws + off);
    off += (((size_t)elems * 4 + 255) / 256) * 256;
    return p;
  };
  float* gsa1  = alloc(33554432);   // [2,256,65536] — region reused later
  float* aT    = alloc(16777216);   // [2,128,65536] — region reused later
  float* meanA = alloc(256);
  float* covB  = alloc(32768);
  float* rcB   = alloc(1024);
  float* attB  = alloc(256);
  float* xg    = alloc(13500416);   // [2,103,65536] — region reused later
  float* x1    = alloc(8388608);
  float* x2    = alloc(4194304);
  float* x3    = alloc(2097152);
  float* x1se  = alloc(8388608);
  float* x2se  = alloc(4194304);
  float* x4    = alloc(4194304);
  float* avgB  = alloc(128);
  // aliased temporaries (lifetimes verified):
  float* g1x2u = gsa1;              float* g1h  = gsa1 + 8388608;
  float* g1g1  = gsa1 + 16777216;   float* g1g2 = gsa1 + 16908288;
  float* g1x1  = gsa1 + 17039360;
  float* g1cat = aT;                                    // 16777216 exact
  float* g2x1  = gsa1;              float* g2t  = gsa1 + 4194304;
  float* g2x2u = gsa1 + 5242880;    float* g2h  = gsa1 + 9437184;
  float* g2g1  = gsa1 + 13631488;   float* g2g2 = gsa1 + 13664256;
  float* g2cat = gsa1 + 13697024;
  float* xb = xg;                   float* tE = xg + 2097152;
  float* s1 = xg + 2621440;         float* u1 = xg + 4718592;
  float* s2  = aT;
  float* u2b = gsa1;                float* x5 = gsa1 + 16777216;
  float* s3  = xg;

  auto conv = [&](const float* inp, const float* w, const float* bi, const float* ga,
                  const float* be, float* o, int Ci, int Co, int H, int W, int Ho, int Wo,
                  int KH, int KW, int st, int pd, int dl, int ibs, int obs,
                  int act, int post) {
    dim3 g((unsigned)(((long)Ho * Wo + 63) / 64), (unsigned)((Co + 63) / 64), 2);
    wmma_conv_kernel<<<g, 256, 0, stream>>>(inp, w, bi, ga, be, o, Ci, Co, H, W, Ho, Wo,
                                            KH, KW, st, pd, dl, ibs, obs, act, post);
  };
  auto blocks = [](long n) { return dim3((unsigned)((n + 255) / 256)); };
  auto addl = [&](const float* A, long as, const float* Bp, long bs,
                  float* O, long os, long perB) {
    add_kernel<<<blocks(2 * perB), 256, 0, stream>>>(A, as, Bp, bs, O, os, perB);
  };
  auto up2l = [&](const float* inp, float* o, int C, int H, int W) {
    long n = 2L * C * 4 * H * W;
    up2_kernel<<<blocks(n), 256, 0, stream>>>(inp, o, C, H, W, n);
  };

  const int P256 = 65536, P128 = 16384, P64 = 4096;

  // ================= GSA =================
  conv(X, gsa_c1_w, nullptr, gsa_bn1_g, gsa_bn1_b, gsa1,
       103, 256, 256, 256, 256, 256, 1, 1, 1, 0, 1, 103 * P256, 256 * P256, ACT_RELU, 0);
  conv(gsa1, gsa_dr_w, gsa_dr_b, gsa_dr_g, gsa_dr_bb, aT,
       256, 128, 256, 256, 256, 256, 1, 1, 1, 0, 1, 256 * P256, 128 * P256, ACT_RELU, 0);
  mean_hw_kernel<<<dim3(256), 256, 0, stream>>>(aT, meanA, 65536);
  { dim3 g(2, 2, 2);
    wmma_covpool_kernel<<<g, 256, 0, stream>>>(aT, meanA, gsa_row_g, gsa_row_b, covB, 128, 65536); }
  rc_kernel<<<dim3(256), 128, 0, stream>>>(covB, gsa_rc_w, gsa_rc_b, rcB, 128);
  conv(rcB, gsa_fc_w, gsa_fc_b, nullptr, nullptr, attB,
       512, 103, 1, 1, 1, 1, 1, 1, 1, 0, 1, 512, 103, ACT_SIG, 0);
  { long n = 2L * 103 * P256;
    scale_att_kernel<<<blocks(n), 256, 0, stream>>>(X, attB, xg, 103, P256, n); }

  // ================= backbone =================
  conv(xg, c1_w, c1_b, c1_g, c1_bb, x1, 103, 64, 256, 256, 256, 256,
       3, 3, 1, 1, 1, 103 * P256, 64 * P256, ACT_RELU, 0);
  conv(x1, c2_w, c2_b, c2_g, c2_bb, x2, 64, 128, 256, 256, 128, 128,
       3, 3, 2, 1, 1, 64 * P256, 128 * P128, ACT_RELU, 0);
  conv(x2, c3_w, c3_b, c3_g, c3_bb, x3, 128, 256, 128, 128, 64, 64,
       3, 3, 2, 1, 1, 128 * P128, 256 * P64, ACT_RELU, 0);

  // ================= GSSI 1 (x1, x2 -> x1se @256, C=64) =================
  conv(x1, g1_c1_w, nullptr, g1_c1_g, g1_c1_b, g1x1, 64, 64, 256, 256, 256, 256,
       1, 1, 1, 0, 1, 64 * P256, 64 * P256, ACT_RELU, 0);
  conv(x2, g1_c2_w, nullptr, g1_c2_g, g1_c2_b, g1h /*temp t*/, 128, 64, 128, 128, 128, 128,
       1, 1, 1, 0, 1, 128 * P128, 64 * P128, ACT_RELU, 0);
  up2l(g1h, g1x2u, 64, 128, 128);
  conv(g1x1, g1_a_w1, g1_a_b1, g1_a_g, g1_a_bb, g1h, 64, 64, 256, 256, 256, 256,
       1, 1, 1, 0, 1, 64 * P256, 64 * P256, ACT_RELU, 1);            // bn(relu(.)): post-affine
  conv(g1h, g1_a_w2, g1_a_b2, g1_a_g2, g1_a_bb2, g1g1, 64, 1, 256, 256, 256, 256,
       1, 1, 1, 0, 1, 64 * P256, P256, ACT_SIG, 1);                  // bn(sigmoid(.))
  conv(g1x2u, g1_b_w1, g1_b_b1, g1_b_g, g1_b_bb, g1h, 64, 64, 256, 256, 256, 256,
       1, 1, 1, 0, 1, 64 * P256, 64 * P256, ACT_RELU, 1);
  conv(g1h, g1_b_w2, g1_b_b2, g1_b_g2, g1_b_bb2, g1g2, 64, 1, 256, 256, 256, 256,
       1, 1, 1, 0, 1, 64 * P256, P256, ACT_SIG, 1);
  { long n = 2L * 128 * P256;
    gff_kernel<<<blocks(n), 256, 0, stream>>>(g1x1, g1x2u, g1g1, g1g2, g1cat, 64, P256, n); }
  conv(g1cat, g1_p_w, nullptr, g1_p_g, g1_p_b, x1se, 128, 64, 256, 256, 256, 256,
       1, 1, 1, 0, 1, 128 * P256, 64 * P256, ACT_RELU, 0);

  // ================= GSSI 2 (x2, x3 -> x2se @128, C=128) =================
  conv(x2, g2_c1_w, nullptr, g2_c1_g, g2_c1_b, g2x1, 128, 128, 128, 128, 128, 128,
       1, 1, 1, 0, 1, 128 * P128, 128 * P128, ACT_RELU, 0);
  conv(x3, g2_c2_w, nullptr, g2_c2_g, g2_c2_b, g2t, 256, 128, 64, 64, 64, 64,
       1, 1, 1, 0, 1, 256 * P64, 128 * P64, ACT_RELU, 0);
  up2l(g2t, g2x2u, 128, 64, 64);
  conv(g2x1, g2_a_w1, g2_a_b1, g2_a_g, g2_a_bb, g2h, 128, 128, 128, 128, 128, 128,
       1, 1, 1, 0, 1, 128 * P128, 128 * P128, ACT_RELU, 1);
  conv(g2h, g2_a_w2, g2_a_b2, g2_a_g2, g2_a_bb2, g2g1, 128, 1, 128, 128, 128, 128,
       1, 1, 1, 0, 1, 128 * P128, P128, ACT_SIG, 1);
  conv(g2x2u, g2_b_w1, g2_b_b1, g2_b_g, g2_b_bb, g2h, 128, 128, 128, 128, 128, 128,
       1, 1, 1, 0, 1, 128 * P128, 128 * P128, ACT_RELU, 1);
  conv(g2h, g2_b_w2, g2_b_b2, g2_b_g2, g2_b_bb2, g2g2, 128, 1, 128, 128, 128, 128,
       1, 1, 1, 0, 1, 128 * P128, P128, ACT_SIG, 1);
  { long n = 2L * 256 * P128;
    gff_kernel<<<blocks(n), 256, 0, stream>>>(g2x1, g2x2u, g2g1, g2g2, g2cat, 128, P128, n); }
  conv(g2cat, g2_p_w, nullptr, g2_p_g, g2_p_b, x2se, 256, 128, 128, 128, 128, 128,
       1, 1, 1, 0, 1, 256 * P128, 128 * P128, ACT_RELU, 0);

  // ================= EMFF on x3 [2,256,64,64] -> xb =================
  conv(x3, e1_w, nullptr, e1_g, e1_b, xb, 64, 64, 64, 64, 64, 64,
       3, 3, 1, 1, 1, 256 * P64, 256 * P64, ACT_RELU, 0);                 // b1 -> slice 0
  addl(x3 + 64 * P64, 256 * P64, xb, 256 * P64, tE, 64 * P64, 64 * P64);  // x1_ + b1
  conv(tE, e2_w, nullptr, e2_g, e2_b, xb + 64 * P64, 64, 64, 64, 64, 64, 64,
       3, 3, 1, 2, 2, 64 * P64, 256 * P64, ACT_RELU, 0);                  // b2 -> slice 1
  addl(x3 + 128 * P64, 256 * P64, xb + 64 * P64, 256 * P64, tE, 64 * P64, 64 * P64);
  conv(tE, e3_w, nullptr, e3_g, e3_b, xb + 128 * P64, 64, 64, 64, 64, 64, 64,
       3, 3, 1, 3, 3, 64 * P64, 256 * P64, ACT_RELU, 0);                  // b3 -> slice 2
  addl(x3 + 192 * P64, 256 * P64, xb + 128 * P64, 256 * P64, tE, 64 * P64, 64 * P64);
  mean_hw_kernel<<<dim3(128), 256, 0, stream>>>(tE, avgB, 4096);
  { long n = 2L * 64 * P64;
    bcast_avg_kernel<<<blocks(n), 256, 0, stream>>>(avgB, eavg_g, eavg_b, xb + 192 * P64,
                                                    256 * P64, 64, P64, n); }

  // ================= decoder =================
  addl(xb, 256 * P64, x3, 256 * P64, s1, 256 * P64, 256 * P64);
  up2l(s1, u1, 256, 64, 64);
  conv(u1, c4_w, c4_b, c4_g, c4_bb, x4, 256, 128, 128, 128, 128, 128,
       3, 3, 1, 1, 1, 256 * P128, 128 * P128, ACT_RELU, 0);
  addl(x4, 128 * P128, x2se, 128 * P128, s2, 128 * P128, 128 * P128);
  up2l(s2, u2b, 128, 128, 128);
  conv(u2b, c5_w, c5_b, c5_g, c5_bb, x5, 128, 64, 256, 256, 256, 256,
       3, 3, 1, 1, 1, 128 * P256, 64 * P256, ACT_RELU, 0);
  addl(x5, 64 * P256, x1se, 64 * P256, s3, 64 * P256, 64 * P256);
  conv(s3, last_w, last_b, nullptr, nullptr, (float*)d_out, 64, 9, 256, 256, 256, 256,
       1, 1, 1, 0, 1, 64 * P256, 9 * P256, ACT_NONE, 0);
}
